// LCBCAdapter_18511309046170
// MI455X (gfx1250) — matrix operand order repaired and checked by reading, not compiled, not run on hardware
//
#include <hip/hip_runtime.h>
#include <math.h>

typedef __bf16 bf16;
typedef __attribute__((ext_vector_type(16))) __bf16        v16bf;
typedef __attribute__((ext_vector_type(8)))  float         v8f;
typedef __attribute__((ext_vector_type(8)))  unsigned int  v8u;

#define NPIX 4096
#define CCH  256
#define EDIM 96
#define NBAT 4
#define ATT_SCALE 1.0206207261596576f /* 10/sqrt(96) */

// ---------------- WMMA layout helpers (CDNA5 ISA 7.12.2, wave32) ----------------
// All fragment loads fetch contiguous bf16 pairs as u32 (8 dword loads per fragment).

// A 16x32 (16-bit): lane m=l%16; lanes<16: V0..3 K=0..7, V4..7 K=16..23;
//                   lanes>=16: V0..3 K=8..15, V4..7 K=24..31.  src is [M][ld] row-major.
static __device__ __forceinline__ v16bf load_A16x32(const bf16* src, int ld, int lane) {
  const int m  = lane & 15;
  const int kb = (lane < 16) ? 0 : 8;
  const bf16* row = src + (size_t)m * ld;
  v8u t;
#pragma unroll
  for (int j = 0; j < 8; ++j) {
    const int k = ((j >= 4) ? 16 : 0) + kb + 2 * (j & 3);
    t[j] = *(const unsigned int*)(row + k);
  }
  return __builtin_bit_cast(v16bf, t);
}

// B 32x16 (16-bit): lane col n=l%16; lanes<16: V0..3 K=0..7, V4..7 K=16..23;
//                   lanes>=16: V0..3 K=8..15, V4..7 K=24..31 (matches A).
// Memory holds B^T as [N][ld] row-major -> per lane K runs contiguously.
static __device__ __forceinline__ v16bf load_B32x16(const bf16* src, int ld, int lane) {
  const int n  = lane & 15;
  const int kb = (lane < 16) ? 0 : 8;
  const bf16* row = src + (size_t)n * ld;
  v8u t;
#pragma unroll
  for (int j = 0; j < 8; ++j)
    t[j] = *(const unsigned int*)(row + ((j >= 4) ? 16 : 0) + kb + 2 * (j & 3));
  return __builtin_bit_cast(v16bf, t);
}

static __device__ __forceinline__ v8f wmma_bf16(v16bf a, v16bf b, v8f c) {
  return __builtin_amdgcn_wmma_f32_16x16x32_bf16(false, a, false, b, (short)0, c, false, false);
}

// ---------------- weight convert (f32 -> bf16, optional zero-pad cols) ----------------
__global__ void convert_pad_k(const float* __restrict__ src, bf16* __restrict__ dst,
                              int rows, int scols, int dcols) {
  const int i = blockIdx.x * 256 + threadIdx.x;
  if (i >= rows * dcols) return;
  const int r = i / dcols, c = i - r * dcols;
  dst[i] = (c < scols) ? (bf16)src[(size_t)r * scols + c] : (bf16)0.0f;
}

// ---------------- channel LayerNorm (thread per pixel) ----------------
// Writes xn32 channel-major (f32, reused later) and xnb/xb PIXEL-major bf16
// ([b][pixel][C]) so GEMM B fragments are contiguous-pair loads.
__global__ void ln_k(const float* __restrict__ x, const float* __restrict__ gamma,
                     const float* __restrict__ beta, float* __restrict__ xn32,
                     bf16* __restrict__ xnb, bf16* __restrict__ xb) {
  const int p = blockIdx.x * 256 + threadIdx.x;
  if (p >= NBAT * NPIX) return;
  const int b = p >> 12, pp = p & (NPIX - 1);
  const float* xp = x + (size_t)b * CCH * NPIX + pp;
  float s = 0.f, ss = 0.f;
  for (int c = 0; c < CCH; ++c) {
    const float v = xp[(size_t)c * NPIX];
    s += v; ss += v * v;
  }
  const float mu = s * (1.0f / CCH);
  const float var = ss * (1.0f / CCH) - mu * mu;
  const float rstd = rsqrtf(var + 1e-5f);
  bf16* xnbp = xnb + (size_t)p * CCH;
  bf16* xbp  = xb  + (size_t)p * CCH;
  for (int c = 0; c < CCH; ++c) {
    const float v = xp[(size_t)c * NPIX];
    const float xn = (v - mu) * rstd * gamma[c] + beta[c];
    xn32[((size_t)b * CCH + c) * NPIX + pp] = xn;
    xnbp[c] = (bf16)xn;
    xbp[c]  = (bf16)v;
  }
}

// ---------------- cosine-normalize Q (with w_qm*mask) / K -> [pixel][96] bf16 ----------------
__global__ void qk_norm_k(const float* __restrict__ raw, const float* __restrict__ w_qm,
                          const float* __restrict__ mask, bf16* __restrict__ out, int use_mask) {
  const int p = blockIdx.x * 256 + threadIdx.x;
  if (p >= NBAT * NPIX) return;
  const int b = p >> 12, pp = p & (NPIX - 1);
  const float mval = use_mask ? mask[p] : 0.f;
  float v[EDIM];
  float ss = 0.f;
#pragma unroll
  for (int e = 0; e < EDIM; ++e) {
    float t = raw[((size_t)b * EDIM + e) * NPIX + pp];
    if (use_mask) t += w_qm[e] * mval;
    v[e] = t; ss += t * t;
  }
  const float inv = 1.0f / fmaxf(sqrtf(ss), 1e-12f);
#pragma unroll
  for (int e = 0; e < EDIM; ++e)
    out[(size_t)p * EDIM + e] = (bf16)(v[e] * inv);
}

// ---------------- small elementwise preps (pixel-major destinations) ----------------
__global__ void prep_cin2_k(const float* __restrict__ xn32, const float* __restrict__ mask,
                            bf16* __restrict__ cin) {
  const int i = blockIdx.x * 256 + threadIdx.x;
  if (i >= NBAT * CCH * NPIX) return;
  const int b = i / (CCH * NPIX);
  const int c = (i / NPIX) % CCH;
  const int p = i & (NPIX - 1);
  cin[((size_t)b * NPIX + p) * 512 + CCH + c] =
      (bf16)(xn32[i] * mask[(size_t)b * NPIX + p]);
}

__global__ void prep_gin_extra_k(const float* __restrict__ mask, bf16* __restrict__ gin) {
  const int i = blockIdx.x * 256 + threadIdx.x;
  if (i >= NBAT * 32 * NPIX) return;
  const int b = i / (32 * NPIX);
  const int r = (i / NPIX) % 32;
  const int p = i & (NPIX - 1);
  gin[((size_t)b * NPIX + p) * 288 + CCH + r] =
      (r == 0) ? (bf16)mask[(size_t)b * NPIX + p] : (bf16)0.0f;
}

// ---------------- generic WMMA GEMM: OUT[M,N] = W[M,K] x X[K,N] per batch ----------------
// X is stored transposed (pixel-major [N][xld]) so B fragments are contiguous pairs.
// One wave per 16(M)x64(N) tile.  EPI selects the fused epilogue.
template <int EPI>
__global__ __launch_bounds__(32, 1) void gemm_k(
    const bf16* __restrict__ W, const bf16* __restrict__ X,
    int M, int K, int xld, size_t ostride,
    float* __restrict__ outf, bf16* __restrict__ outb,
    const float* __restrict__ aux0, const float* __restrict__ aux1,
    const float* __restrict__ aux2, const float* __restrict__ aux3) {
  const int b   = blockIdx.z;
  const int m0  = blockIdx.y * 16;
  const int n0  = blockIdx.x * 64;
  const int lane = threadIdx.x;
  const bf16* Wt = W + (size_t)m0 * K;
  const bf16* Xb = X + ((size_t)b * NPIX + n0) * xld;

  const v8f zero = {0.f, 0.f, 0.f, 0.f, 0.f, 0.f, 0.f, 0.f};
  v8f acc[4] = {zero, zero, zero, zero};

  for (int k0 = 0; k0 < K; k0 += 32) {
    const v16bf a = load_A16x32(Wt + k0, K, lane);
#pragma unroll
    for (int j = 0; j < 4; ++j) {
      const v16bf bm = load_B32x16(Xb + (size_t)(j * 16) * xld + k0, xld, lane);
      acc[j] = wmma_bf16(a, bm, acc[j]);
    }
    if (k0 + 32 < K)
      __builtin_prefetch((const void*)(Wt + k0 + 32), 0, 0);
  }

  const int half = lane >> 4, lc = lane & 15;
#pragma unroll
  for (int j = 0; j < 4; ++j) {
#pragma unroll
    for (int r = 0; r < 8; ++r) {
      const int m = m0 + r + 8 * half;
      const int n = n0 + j * 16 + lc;
      const float v = acc[j][r];
      const size_t iCN = ((size_t)b * CCH + m) * NPIX + n;
      if constexpr (EPI == 0) {           // plain f32 [b][M][N] (q/k raw)
        outf[(size_t)b * ostride + (size_t)m * NPIX + n] = v;
      } else if constexpr (EPI == 1) {    // v proj: bf16 channel-major [b][C][pixel]
        outb[(size_t)b * ostride + (size_t)m * NPIX + n] = (bf16)v;
      } else if constexpr (EPI == 2) {    // w_out: corr_in[0:256] = attended - x (pixel-major)
        outb[((size_t)b * NPIX + n) * 512 + m] = (bf16)(v - aux0[iCN]);
      } else if constexpr (EPI == 3) {    // wd1: exact gelu -> hidden (pixel-major)
        const float g = 0.5f * v * (1.0f + erff(v * 0.70710678118f));
        outb[((size_t)b * NPIX + n) * 256 + m] = (bf16)g;
      } else if constexpr (EPI == 4) {    // wd2: corrective f32 + bf16 gate input (pixel-major)
        outf[iCN] = v;
        outb[((size_t)b * NPIX + n) * 288 + m] = (bf16)v;
      } else {                            // wg: out = x + tanh(rs)*sigmoid(.)*mask*corr
        const float mval = aux1[(size_t)b * NPIX + n];
        const float gate = (1.0f / (1.0f + expf(-v))) * mval;
        const float rsv  = tanhf(aux2[0]);
        outf[iCN] = aux3[iCN] + rsv * gate * aux0[iCN];
      }
    }
  }
}

// ---------------- fused flash attention ----------------
// Block: 256 threads = 8 waves, one 16-query tile per block.
// Q tile staged in LDS (kills the VGPR spill of Q fragments across the key loop).
// wave0: S = Q K^T (6 WMMA / 32-key chunk), mask bias, online softmax, P -> LDS.
// all 8 waves: 32-channel slice of P x V (2 WMMA / chunk).
__global__ __launch_bounds__(256, 1) void flash_attn_k(
    const bf16* __restrict__ qb, const bf16* __restrict__ kbuf,
    const bf16* __restrict__ vb, const float* __restrict__ mask,
    bf16* __restrict__ attb) {
  const int b  = blockIdx.y;
  const int q0 = blockIdx.x * 16;
  const int tid  = threadIdx.x;
  const int w    = tid >> 5;
  const int lane = tid & 31;
  const int half = lane >> 4;
  const int lc   = lane & 15;

  __shared__ bf16  q_lds[16 * EDIM];
  __shared__ bf16  p_lds[16 * 32];
  __shared__ float scale_lds[16];
  __shared__ float invl_lds[16];

  // cooperative copy of the 16x96 Q tile into LDS (768 dwords, 3 per thread)
  {
    const unsigned int* qg = (const unsigned int*)(qb + ((size_t)b * NPIX + q0) * EDIM);
    unsigned int* ql = (unsigned int*)q_lds;
#pragma unroll
    for (int j = 0; j < 3; ++j)
      ql[tid + j * 256] = qg[tid + j * 256];
  }
  __syncthreads();

  const v8f zero = {0.f, 0.f, 0.f, 0.f, 0.f, 0.f, 0.f, 0.f};
  v8f acc0 = zero, acc1 = zero;

  float mi[8], li[8];
#pragma unroll
  for (int r = 0; r < 8; ++r) { mi[r] = -3.0e38f; li[r] = 0.f; }
  const int c0 = w * 32;

  for (int k0 = 0; k0 < NPIX; k0 += 32) {
    if (w == 0) {
      // Q fragments re-materialized from LDS each iteration (cheap ds loads)
      const v16bf qa0 = load_A16x32(q_lds,      EDIM, lane);
      const v16bf qa1 = load_A16x32(q_lds + 32, EDIM, lane);
      const v16bf qa2 = load_A16x32(q_lds + 64, EDIM, lane);
      v8f s0 = zero, s1 = zero;
      const bf16* kb0 = kbuf + ((size_t)b * NPIX + k0) * EDIM;
      const bf16* kb1 = kb0 + 16 * EDIM;
      s0 = wmma_bf16(qa0, load_B32x16(kb0,      EDIM, lane), s0);
      s0 = wmma_bf16(qa1, load_B32x16(kb0 + 32, EDIM, lane), s0);
      s0 = wmma_bf16(qa2, load_B32x16(kb0 + 64, EDIM, lane), s0);
      s1 = wmma_bf16(qa0, load_B32x16(kb1,      EDIM, lane), s1);
      s1 = wmma_bf16(qa1, load_B32x16(kb1 + 32, EDIM, lane), s1);
      s1 = wmma_bf16(qa2, load_B32x16(kb1 + 64, EDIM, lane), s1);

      const float inv0 = mask[(size_t)b * NPIX + k0 + lc];
      const float inv1 = mask[(size_t)b * NPIX + k0 + 16 + lc];
      float chmax[8];
#pragma unroll
      for (int r = 0; r < 8; ++r) {
        float a = s0[r] * ATT_SCALE; if (inv0 >= 0.5f) a = -10000.f;
        float c = s1[r] * ATT_SCALE; if (inv1 >= 0.5f) c = -10000.f;
        s0[r] = a; s1[r] = c;
        chmax[r] = fmaxf(a, c);
      }
#pragma unroll
      for (int sh = 1; sh <= 8; sh <<= 1)
#pragma unroll
        for (int r = 0; r < 8; ++r)
          chmax[r] = fmaxf(chmax[r], __shfl_xor(chmax[r], sh, 32));

      float rowsum[8], fscale[8];
#pragma unroll
      for (int r = 0; r < 8; ++r) {
        const float nm = fmaxf(mi[r], chmax[r]);
        fscale[r] = expf(mi[r] - nm);
        const float p0 = expf(s0[r] - nm);
        const float p1 = expf(s1[r] - nm);
        s0[r] = p0; s1[r] = p1;
        rowsum[r] = p0 + p1;
        mi[r] = nm;
      }
#pragma unroll
      for (int sh = 1; sh <= 8; sh <<= 1)
#pragma unroll
        for (int r = 0; r < 8; ++r)
          rowsum[r] += __shfl_xor(rowsum[r], sh, 32);
#pragma unroll
      for (int r = 0; r < 8; ++r)
        li[r] = li[r] * fscale[r] + rowsum[r];

#pragma unroll
      for (int r = 0; r < 8; ++r) {
        const int row = r + 8 * half;
        p_lds[row * 32 + lc]      = (bf16)s0[r];
        p_lds[row * 32 + 16 + lc] = (bf16)s1[r];
      }
      if (lc == 0) {
#pragma unroll
        for (int r = 0; r < 8; ++r) scale_lds[r + 8 * half] = fscale[r];
      }
    }
    __syncthreads();

    // every wave: rescale its accumulator, then P x V for its 32-channel slice
#pragma unroll
    for (int r = 0; r < 8; ++r) {
      const float f = scale_lds[r + 8 * half];
      acc0[r] *= f; acc1[r] *= f;
    }
    const v16bf pa = load_A16x32(p_lds, 32, lane);
    // V is channel-major [b][C][key] -> B^T layout, contiguous key pairs per lane
    const bf16* vbase = vb + ((size_t)b * CCH + c0) * NPIX + k0;
    acc0 = wmma_bf16(pa, load_B32x16(vbase,                    NPIX, lane), acc0);
    acc1 = wmma_bf16(pa, load_B32x16(vbase + (size_t)16 * NPIX, NPIX, lane), acc1);
    __syncthreads();
  }

  if (w == 0 && lc == 0) {
#pragma unroll
    for (int r = 0; r < 8; ++r) invl_lds[r + 8 * half] = 1.0f / li[r];
  }
  __syncthreads();

  // store attended pixel-major [b][pixel][C]: contiguous across lanes
#pragma unroll
  for (int r = 0; r < 8; ++r) {
    const float f = invl_lds[r + 8 * half];
    const int pix = q0 + r + 8 * half;
    bf16* dst = attb + ((size_t)b * NPIX + pix) * CCH;
    dst[c0 + lc]      = (bf16)(acc0[r] * f);
    dst[c0 + 16 + lc] = (bf16)(acc1[r] * f);
  }
}

// ---------------- host orchestration ----------------
extern "C" void kernel_launch(void* const* d_in, const int* in_sizes, int n_in,
                              void* d_out, int out_size, void* d_ws, size_t ws_size,
                              hipStream_t stream) {
  const float* x    = (const float*)d_in[0];
  const float* mask = (const float*)d_in[1];
  const float* gam  = (const float*)d_in[2];
  const float* bet  = (const float*)d_in[3];
  const float* wq   = (const float*)d_in[4];
  const float* wk   = (const float*)d_in[5];
  const float* wv   = (const float*)d_in[6];
  const float* wout = (const float*)d_in[7];
  const float* wqm  = (const float*)d_in[8];
  const float* wd1  = (const float*)d_in[9];
  const float* wd2  = (const float*)d_in[10];
  const float* wg   = (const float*)d_in[11];
  const float* rs   = (const float*)d_in[12];
  float* out = (float*)d_out;

  char* ws = (char*)d_ws;
  size_t off = 0;
  auto alloc = [&](size_t bytes) -> void* {
    void* p = ws + off;
    off += (bytes + 255) & ~(size_t)255;
    return p;
  };

  const size_t CN = (size_t)CCH * NPIX;
  bf16* wqb  = (bf16*)alloc((size_t)EDIM * CCH * 2);
  bf16* wkb  = (bf16*)alloc((size_t)EDIM * CCH * 2);
  bf16* wvb  = (bf16*)alloc((size_t)CCH * CCH * 2);
  bf16* wob  = (bf16*)alloc((size_t)CCH * CCH * 2);
  bf16* wd1b = (bf16*)alloc((size_t)CCH * 512 * 2);
  bf16* wd2b = (bf16*)alloc((size_t)CCH * CCH * 2);
  bf16* wgb  = (bf16*)alloc((size_t)CCH * 288 * 2);
  float* xn32 = (float*)alloc((size_t)NBAT * CN * 4);          // channel-major
  bf16* xnb   = (bf16*)alloc((size_t)NBAT * CN * 2);           // pixel-major
  bf16* xbb   = (bf16*)alloc((size_t)NBAT * CN * 2);           // pixel-major
  float* qraw = (float*)alloc((size_t)NBAT * EDIM * NPIX * 4); // channel-major
  float* kraw = (float*)alloc((size_t)NBAT * EDIM * NPIX * 4);
  bf16* qb    = (bf16*)alloc((size_t)NBAT * NPIX * EDIM * 2);  // [pixel][96]
  bf16* kb    = (bf16*)alloc((size_t)NBAT * NPIX * EDIM * 2);  // [pixel][96]
  bf16* vb    = (bf16*)alloc((size_t)NBAT * CN * 2);           // channel-major [C][key]
  bf16* attb  = (bf16*)alloc((size_t)NBAT * CN * 2);           // pixel-major
  bf16* cin   = (bf16*)alloc((size_t)NBAT * NPIX * 512 * 2);   // pixel-major [p][512]
  bf16* hid   = (bf16*)alloc((size_t)NBAT * CN * 2);           // pixel-major
  float* corr = (float*)alloc((size_t)NBAT * CN * 4);          // channel-major
  bf16* gin   = (bf16*)alloc((size_t)NBAT * NPIX * 288 * 2);   // pixel-major [p][288]

  auto cvt = [&](const float* s, bf16* d, int rows, int sc, int dc) {
    const int tot = rows * dc;
    convert_pad_k<<<(tot + 255) / 256, 256, 0, stream>>>(s, d, rows, sc, dc);
  };
  cvt(wq,   wqb,  EDIM, CCH, CCH);
  cvt(wk,   wkb,  EDIM, CCH, CCH);
  cvt(wv,   wvb,  CCH,  CCH, CCH);
  cvt(wout, wob,  CCH,  CCH, CCH);
  cvt(wd1,  wd1b, CCH,  512, 512);
  cvt(wd2,  wd2b, CCH,  CCH, CCH);
  cvt(wg,   wgb,  CCH,  257, 288);   // zero-pad K to 288 (multiple of 32)

  ln_k<<<(NBAT * NPIX + 255) / 256, 256, 0, stream>>>(x, gam, bet, xn32, xnb, xbb);

  const dim3 blk(32);
  // Q/K raw projections (M=96), X = xnb pixel-major (xld=256)
  gemm_k<0><<<dim3(64, 6, NBAT), blk, 0, stream>>>(wqb, xnb, EDIM, CCH, CCH, (size_t)EDIM * NPIX,
                                                   qraw, nullptr, nullptr, nullptr, nullptr, nullptr);
  gemm_k<0><<<dim3(64, 6, NBAT), blk, 0, stream>>>(wkb, xnb, EDIM, CCH, CCH, (size_t)EDIM * NPIX,
                                                   kraw, nullptr, nullptr, nullptr, nullptr, nullptr);
  // V projection -> channel-major bf16 [b][C][key]
  gemm_k<1><<<dim3(64, 16, NBAT), blk, 0, stream>>>(wvb, xbb, CCH, CCH, CCH, CN,
                                                    nullptr, vb, nullptr, nullptr, nullptr, nullptr);
  // cosine normalize
  qk_norm_k<<<(NBAT * NPIX + 255) / 256, 256, 0, stream>>>(qraw, wqm, mask, qb, 1);
  qk_norm_k<<<(NBAT * NPIX + 255) / 256, 256, 0, stream>>>(kraw, wqm, mask, kb, 0);

  // fused flash attention -> attb pixel-major
  flash_attn_k<<<dim3(NPIX / 16, NBAT), 256, 0, stream>>>(qb, kb, vb, mask, attb);

  // w_out + (attended - x) into corr_in[0:256] (pixel-major, xld=256)
  gemm_k<2><<<dim3(64, 16, NBAT), blk, 0, stream>>>(wob, attb, CCH, CCH, CCH, 0,
                                                    nullptr, cin, x, nullptr, nullptr, nullptr);
  // corr_in[256:512] = x_norm * mask
  prep_cin2_k<<<(int)((NBAT * CN + 255) / 256), 256, 0, stream>>>(xn32, mask, cin);
  // hidden = gelu(wd1 x corr_in), xld=512
  gemm_k<3><<<dim3(64, 16, NBAT), blk, 0, stream>>>(wd1b, cin, CCH, 512, 512, 0,
                                                    nullptr, hid, nullptr, nullptr, nullptr, nullptr);
  // corrective = wd2 x hidden (f32 channel-major + bf16 gate-input pixel-major)
  gemm_k<4><<<dim3(64, 16, NBAT), blk, 0, stream>>>(wd2b, hid, CCH, CCH, CCH, 0,
                                                    corr, gin, nullptr, nullptr, nullptr, nullptr);
  // gate input cols 256..287 (mask col + zero pad)
  prep_gin_extra_k<<<(NBAT * 32 * NPIX + 255) / 256, 256, 0, stream>>>(mask, gin);
  // final: out = x + tanh(rs) * sigmoid(wg x gate_in) * mask * corrective, xld=288
  gemm_k<5><<<dim3(64, 16, NBAT), blk, 0, stream>>>(wgb, gin, CCH, 288, 288, 0,
                                                    out, nullptr, corr, mask, rs, x);

  (void)in_sizes; (void)n_in; (void)out_size; (void)ws_size;
}
